// HGT_72232759984505
// MI455X (gfx1250) — compile-verified, matrix operand order checked by reading, not run on hardware
//
#include <hip/hip_runtime.h>
#include <hip/hip_bf16.h>
#include <math.h>

typedef __attribute__((ext_vector_type(16))) _Float16 v16h;
typedef __attribute__((ext_vector_type(8)))  float    v8f;

#define NA 50000
#define NP 100000
#define DHID 32
#define DFEAT 128
#define EW 500000
#define ER 500000
#define EC 1000000

// ---------------------------------------------------------------------------
// order-preserving float <-> uint encoding for atomicMax-based segment max
// ---------------------------------------------------------------------------
__device__ __forceinline__ unsigned encF(float f) {
    unsigned u = __float_as_uint(f);
    return (u & 0x80000000u) ? ~u : (u | 0x80000000u);
}
__device__ __forceinline__ float decF(unsigned u) {
    return (u & 0x80000000u) ? __uint_as_float(u & 0x7fffffffu)
                             : __uint_as_float(~u);
}
#define NEG_INF_ENC 0x007fffffu   // encF(-inf)

__device__ __forceinline__ float gelu_exact(float x) {
    return 0.5f * x * (1.0f + erff(x * 0.70710678118654752f));
}

// ---------------------------------------------------------------------------
// Row-stripe WMMA GEMM:  C[M,N] = A[M,K] @ W[K,N] (+bias) (+gated skip)
//   NT = N/16 output tiles per wave (full row stripe), KS = K/32 k-steps.
//   W is staged once per block into LDS, pre-swizzled into WMMA B-fragment
//   order so each wave reads one contiguous v16h (2x ds_load_b128) per tile.
//   Requires M%16==0 (50000/100000 both are).
// ---------------------------------------------------------------------------
template<int NT, int KS>
__global__ __launch_bounds__(256)
void wmma_gemm_kernel(const float* __restrict__ A,
                      const float* __restrict__ W,
                      const float* __restrict__ bias,       // nullable
                      const float* __restrict__ skipX,      // nullable
                      const float* __restrict__ skipParam,  // scalar, nullable
                      float* __restrict__ C, int M)
{
    constexpr int N = NT * 16;
    __shared__ _Float16 ldsW[NT * KS * 512];   // [(t*KS+s)*32 + lane]*16 + j

    // ---- cooperative W -> LDS, converted to f16 in B-fragment layout ----
    const int tid = threadIdx.x;
    for (int idx = tid; idx < NT * KS * 512; idx += blockDim.x) {
        int j  = idx & 15;
        int L  = (idx >> 4) & 31;
        int ts = idx >> 9;            // t*KS + s
        int s  = ts % KS;
        int t  = ts / KS;
        int col   = t * 16 + (L & 15);
        int kbase = (L >> 4) * 8;
        int k     = s * 32 + kbase + ((j < 8) ? j : j + 8);
        ldsW[idx] = (_Float16)W[(size_t)k * N + col];
    }
    __syncthreads();

    const int lane   = tid & 31;
    const int wave   = tid >> 5;
    const int stripe = blockIdx.x * (blockDim.x >> 5) + wave;
    if (stripe >= (M >> 4)) return;            // wave-uniform

    const int rlane = lane & 15;
    const int half  = lane >> 4;               // 0 or 1
    const int arow  = stripe * 16 + rlane;     // A row owned by this lane
    const int kbase = half * 8;

    v8f acc[NT];
    #pragma unroll
    for (int t = 0; t < NT; ++t) acc[t] = {};

    const float* Arow = A + (size_t)arow * (KS * 32);
    #pragma unroll
    for (int s = 0; s < KS; ++s) {
        const float* Ap = Arow + s * 32 + kbase;
        float4 x0 = *(const float4*)(Ap);
        float4 x1 = *(const float4*)(Ap + 4);
        float4 x2 = *(const float4*)(Ap + 16);
        float4 x3 = *(const float4*)(Ap + 20);
        v16h a;
        a[0]  = (_Float16)x0.x; a[1]  = (_Float16)x0.y;
        a[2]  = (_Float16)x0.z; a[3]  = (_Float16)x0.w;
        a[4]  = (_Float16)x1.x; a[5]  = (_Float16)x1.y;
        a[6]  = (_Float16)x1.z; a[7]  = (_Float16)x1.w;
        a[8]  = (_Float16)x2.x; a[9]  = (_Float16)x2.y;
        a[10] = (_Float16)x2.z; a[11] = (_Float16)x2.w;
        a[12] = (_Float16)x3.x; a[13] = (_Float16)x3.y;
        a[14] = (_Float16)x3.z; a[15] = (_Float16)x3.w;
        #pragma unroll
        for (int t = 0; t < NT; ++t) {
            v16h b = *(const v16h*)&ldsW[((t * KS + s) * 32 + lane) * 16];
            acc[t] = __builtin_amdgcn_wmma_f32_16x16x32_f16(
                         false, a, false, b, (short)0, acc[t], false, false);
        }
    }

    // ---- epilogue: bias + optional gated skip blend ----
    float al = 1.0f, om = 0.0f;
    if (skipX) {
        float sp = *skipParam;
        al = 1.0f / (1.0f + __expf(-sp));
        om = 1.0f - al;
    }
    const int rbase = stripe * 16 + half * 8;  // D rows = 8*(lane>>4)+i
    #pragma unroll
    for (int t = 0; t < NT; ++t) {
        int ncol = t * 16 + rlane;
        float bn = bias ? bias[ncol] : 0.0f;
        #pragma unroll
        for (int i = 0; i < 8; ++i) {
            int r = rbase + i;
            float val = acc[t][i] + bn;
            if (skipX) val = al * val + om * skipX[(size_t)r * N + ncol];
            C[(size_t)r * N + ncol] = val;
        }
    }
}

// ---------------------------------------------------------------------------
// Elementwise exact GELU (separate pass keeps erff divergence out of GEMM)
// ---------------------------------------------------------------------------
__global__ __launch_bounds__(256)
void gelu_kernel(const float* __restrict__ in, float* __restrict__ out, size_t n)
{
    size_t i = blockIdx.x * (size_t)blockDim.x + threadIdx.x;
    if (i < n) out[i] = gelu_exact(in[i]);
}

// ---------------------------------------------------------------------------
// Pass A: wave-per-edge logits + segment max.   lane = channel (coalesced).
// edges laid out [src[0..E-1], dst[0..E-1]]
// ---------------------------------------------------------------------------
__global__ __launch_bounds__(256)
void edge_logit_max_kernel(const int* __restrict__ edges, int E,
                           const float* __restrict__ qdst,
                           const float* __restrict__ krel,
                           const float* __restrict__ prel, float scale,
                           float* __restrict__ elog,
                           unsigned* __restrict__ mEnc)
{
    int gw   = (blockIdx.x * blockDim.x + threadIdx.x) >> 5;
    int lane = threadIdx.x & 31;
    if (gw >= E) return;
    int src = edges[gw];
    int dst = edges[E + gw];
    float p = qdst[(size_t)dst * 32 + lane] * krel[(size_t)src * 32 + lane];
    #pragma unroll
    for (int off = 16; off > 0; off >>= 1) p += __shfl_xor(p, off, 32);
    float logit = p * prel[0] * scale;
    if (lane == 0) {
        elog[gw] = logit;
        atomicMax(&mEnc[dst], encF(logit));
    }
}

// ---------------------------------------------------------------------------
// Pass B: thread-per-edge  e = exp(logit - m[dst]); s[dst] += e  (in-place)
// ---------------------------------------------------------------------------
__global__ __launch_bounds__(256)
void edge_exp_sum_kernel(const int* __restrict__ edges, int E,
                         const unsigned* __restrict__ mEnc,
                         float* __restrict__ elog,
                         float* __restrict__ ssum)
{
    int e = blockIdx.x * blockDim.x + threadIdx.x;
    if (e >= E) return;
    int dst = edges[E + e];
    float ev = __expf(elog[e] - decF(mEnc[dst]));
    elog[e] = ev;
    atomicAdd(&ssum[dst], ev);
}

// ---------------------------------------------------------------------------
// Pass C: wave-per-edge weighted scatter-add of v_rel rows into agg[dst]
// ---------------------------------------------------------------------------
__global__ __launch_bounds__(256)
void edge_scatter_kernel(const int* __restrict__ edges, int E,
                         const float* __restrict__ elog,
                         const float* __restrict__ ssum,
                         const float* __restrict__ vrel,
                         float* __restrict__ agg)
{
    int gw   = (blockIdx.x * blockDim.x + threadIdx.x) >> 5;
    int lane = threadIdx.x & 31;
    if (gw >= E) return;
    int src = edges[gw];
    int dst = edges[E + gw];
    float w = elog[gw] / (ssum[dst] + 1e-16f);
    atomicAdd(&agg[(size_t)dst * 32 + lane],
              w * vrel[(size_t)src * 32 + lane]);
}

__global__ void fill_u32_kernel(unsigned* __restrict__ p, unsigned v, size_t n)
{
    size_t i = blockIdx.x * (size_t)blockDim.x + threadIdx.x;
    if (i < n) p[i] = v;
}

// ---------------------------------------------------------------------------
// Host orchestration
// ---------------------------------------------------------------------------
struct ConvP {
    const float *kw[2], *kb[2], *qw[2], *qb[2], *vw[2], *vb[2];
    const float *aw[2], *ab[2], *skip[2];
    const float *arel[3], *mrel[3], *prel[3];
};

static void unpack_conv(void* const* d_in, int& idx, ConvP& P)
{
    for (int nt = 0; nt < 2; ++nt) {            // author, paper
        P.kw[nt]   = (const float*)d_in[idx++];
        P.kb[nt]   = (const float*)d_in[idx++];
        P.qw[nt]   = (const float*)d_in[idx++];
        P.qb[nt]   = (const float*)d_in[idx++];
        P.vw[nt]   = (const float*)d_in[idx++];
        P.vb[nt]   = (const float*)d_in[idx++];
        P.aw[nt]   = (const float*)d_in[idx++];
        P.ab[nt]   = (const float*)d_in[idx++];
        P.skip[nt] = (const float*)d_in[idx++];
    }
    for (int r = 0; r < 3; ++r) {               // writes, rev_writes, cites
        P.arel[r] = (const float*)d_in[idx++];
        P.mrel[r] = (const float*)d_in[idx++];
        P.prel[r] = (const float*)d_in[idx++];
    }
}

extern "C" void kernel_launch(void* const* d_in, const int* in_sizes, int n_in,
                              void* d_out, int out_size, void* d_ws, size_t ws_size,
                              hipStream_t stream)
{
    (void)in_sizes; (void)n_in; (void)out_size; (void)ws_size;

    int idx = 0;
    const float* x_author = (const float*)d_in[idx++];
    const float* x_paper  = (const float*)d_in[idx++];
    const int*   e_writes = (const int*)d_in[idx++];
    const int*   e_rev    = (const int*)d_in[idx++];
    const int*   e_cites  = (const int*)d_in[idx++];
    ConvP c1, c2;
    unpack_conv(d_in, idx, c1);
    unpack_conv(d_in, idx, c2);
    const float* dec_aw = (const float*)d_in[idx++];
    const float* dec_ab = (const float*)d_in[idx++];
    const float* dec_pw = (const float*)d_in[idx++];
    const float* dec_pb = (const float*)d_in[idx++];

    // ---- workspace carve-out (floats) ----
    float* ws = (float*)d_ws;
    size_t off = 0;
    auto alloc = [&](size_t n) { float* p = ws + off; off += n; return p; };
    float* K_a  = alloc((size_t)NA * 32);
    float* Q_a  = alloc((size_t)NA * 32);
    float* V_a  = alloc((size_t)NA * 32);
    float* K_p  = alloc((size_t)NP * 32);
    float* Q_p  = alloc((size_t)NP * 32);
    float* V_p  = alloc((size_t)NP * 32);
    float* krel = alloc((size_t)NP * 32);     // sized for largest src type
    float* vrel = alloc((size_t)NP * 32);
    float* agg_a = alloc((size_t)NA * 32);
    float* agg_p = alloc((size_t)NP * 32);
    float* h1_a  = alloc((size_t)NA * 32);
    float* h1_p  = alloc((size_t)NP * 32);
    unsigned* m_a = (unsigned*)alloc(NA);
    float*    s_a = alloc(NA);
    unsigned* m_p = (unsigned*)alloc(NP);
    float*    s_p = alloc(NP);
    float* elog = alloc(EC);                  // max edge count
    float* h2_a = K_a;                        // conv2 K/Q/V dead by node update
    float* h2_p = K_p;

    const float scale = 0.17677669529663687f; // 1/sqrt(32)

    auto gemm = [&](const float* A, const float* W, const float* bias,
                    const float* skipX, const float* skipP, float* C,
                    int M, int N, int K) {
        int stripes = M / 16;
        int blocks  = (stripes + 7) / 8;      // 8 waves / block, 1 stripe/wave
        if (N == 32 && K == 128)
            wmma_gemm_kernel<2, 4><<<blocks, 256, 0, stream>>>(A, W, bias, skipX, skipP, C, M);
        else if (N == 32 && K == 32)
            wmma_gemm_kernel<2, 1><<<blocks, 256, 0, stream>>>(A, W, bias, skipX, skipP, C, M);
        else /* N == 128 && K == 32 */
            wmma_gemm_kernel<8, 1><<<blocks, 256, 0, stream>>>(A, W, bias, skipX, skipP, C, M);
    };
    auto fill = [&](void* p, unsigned v, size_t n) {
        size_t blocks = (n + 255) / 256;
        fill_u32_kernel<<<(unsigned)blocks, 256, 0, stream>>>((unsigned*)p, v, n);
    };
    auto gelu = [&](const float* in, float* out, size_t n) {
        size_t blocks = (n + 255) / 256;
        gelu_kernel<<<(unsigned)blocks, 256, 0, stream>>>(in, out, n);
    };
    auto run_rel = [&](const int* edges, int E,
                       const float* Ksrc, const float* Vsrc, const float* Qdst,
                       const float* arel, const float* mrel, const float* prel,
                       float* aggDst, unsigned* mDst, float* sDst,
                       int Nsrc, int Ndst) {
        gemm(Ksrc, arel, nullptr, nullptr, nullptr, krel, Nsrc, 32, 32);
        gemm(Vsrc, mrel, nullptr, nullptr, nullptr, vrel, Nsrc, 32, 32);
        fill(mDst, NEG_INF_ENC, (size_t)Ndst);
        fill(sDst, 0u, (size_t)Ndst);
        int wblocks = (E + 7) / 8;            // wave-per-edge kernels
        int tblocks = (E + 255) / 256;
        edge_logit_max_kernel<<<wblocks, 256, 0, stream>>>(
            edges, E, Qdst, krel, prel, scale, elog, mDst);
        edge_exp_sum_kernel<<<tblocks, 256, 0, stream>>>(edges, E, mDst, elog, sDst);
        edge_scatter_kernel<<<wblocks, 256, 0, stream>>>(
            edges, E, elog, sDst, vrel, aggDst);
    };
    auto conv_layer = [&](const ConvP& P, const float* Xa, const float* Xp,
                          int din, float* Ha, float* Hp, bool useSkip) {
        gemm(Xa, P.kw[0], P.kb[0], nullptr, nullptr, K_a, NA, 32, din);
        gemm(Xa, P.qw[0], P.qb[0], nullptr, nullptr, Q_a, NA, 32, din);
        gemm(Xa, P.vw[0], P.vb[0], nullptr, nullptr, V_a, NA, 32, din);
        gemm(Xp, P.kw[1], P.kb[1], nullptr, nullptr, K_p, NP, 32, din);
        gemm(Xp, P.qw[1], P.qb[1], nullptr, nullptr, Q_p, NP, 32, din);
        gemm(Xp, P.vw[1], P.vb[1], nullptr, nullptr, V_p, NP, 32, din);
        fill(agg_a, 0u, (size_t)NA * 32);
        fill(agg_p, 0u, (size_t)NP * 32);
        // writes:      author -> paper
        run_rel(e_writes, EW, K_a, V_a, Q_p, P.arel[0], P.mrel[0], P.prel[0],
                agg_p, m_p, s_p, NA, NP);
        // rev_writes:  paper -> author
        run_rel(e_rev, ER, K_p, V_p, Q_a, P.arel[1], P.mrel[1], P.prel[1],
                agg_a, m_a, s_a, NP, NA);
        // cites:       paper -> paper
        run_rel(e_cites, EC, K_p, V_p, Q_p, P.arel[2], P.mrel[2], P.prel[2],
                agg_p, m_p, s_p, NP, NP);
        // node update: Ha = blend( gelu(agg) @ a_w + a_b , X )
        // Q buffers are dead here -> reuse as gelu scratch
        gelu(agg_a, Q_a, (size_t)NA * 32);
        gelu(agg_p, Q_p, (size_t)NP * 32);
        gemm(Q_a, P.aw[0], P.ab[0], useSkip ? Xa : nullptr, P.skip[0], Ha, NA, 32, 32);
        gemm(Q_p, P.aw[1], P.ab[1], useSkip ? Xp : nullptr, P.skip[1], Hp, NP, 32, 32);
    };

    conv_layer(c1, x_author, x_paper, DFEAT, h1_a, h1_p, false); // 128!=32: no skip
    conv_layer(c2, h1_a, h1_p, DHID, h2_a, h2_p, true);          // 32==32: skip

    float* out_a = (float*)d_out;
    float* out_p = out_a + (size_t)NA * DFEAT;
    gemm(h2_a, dec_aw, dec_ab, nullptr, nullptr, out_a, NA, DFEAT, 32);
    gemm(h2_p, dec_pw, dec_pb, nullptr, nullptr, out_p, NP, DFEAT, 32);
}
